// ContextAwareSelfAttention_89386859365028
// MI455X (gfx1250) — matrix operand order repaired and checked by reading, not compiled, not run on hardware
//
#include <hip/hip_runtime.h>
#include <hip/hip_bf16.h>

typedef _Float16 v16h __attribute__((ext_vector_type(16)));
typedef float    v8f  __attribute__((ext_vector_type(8)));

#define NB  32     // batch
#define NT  128    // scan length
#define NS  64     // sentences per step (GEMM M)
#define NF  256    // features (GEMM N and K)
#define XPAD 264   // f16 LDS row stride: 528B rows -> 16B aligned, conflict-free frags
#define EPSF 1e-7f

// W fragment arrays: 8 ksteps * 16 ntiles * 2 halves * 32 lanes * 4 dwords
#define WFRAG_DW (8 * 16 * 2 * 32 * 4)   // 32768 dwords = 128 KB per matrix

__device__ __forceinline__ float wred32(float v) {
#pragma unroll
    for (int o = 16; o > 0; o >>= 1) v += __shfl_xor(v, o, 32);
    return v;
}
__device__ __forceinline__ float wred16(float v) {
#pragma unroll
    for (int o = 8; o > 0; o >>= 1) v += __shfl_xor(v, o, 16);
    return v;
}

// ---------------------------------------------------------------------------
// Prep: decompose W (f32 256x256) into f16 hi/lo fragments in B-operand lane
// order for V_WMMA_F32_16X16X32_F16.
// dword index d = (((kk*16 + nt)*2 + v2)*32 + lane)*4 + dw ; v = v2*4 + dw
// lane half h = lane>>4 ; VGPR v holds K = kk*32 + 16h + 2v, 2v+1 ; n = nt*16 + (lane&15)
// ---------------------------------------------------------------------------
__global__ void prep_w_frags(const float* __restrict__ Wg,
                             unsigned int* __restrict__ whi,
                             unsigned int* __restrict__ wlo) {
    int d = blockIdx.x * blockDim.x + threadIdx.x;
    if (d >= WFRAG_DW) return;
    int dw   = d & 3;
    int lane = (d >> 2) & 31;
    int v2   = (d >> 7) & 1;
    int ntkk = d >> 8;
    int nt   = ntkk & 15;
    int kk   = ntkk >> 4;
    int v    = v2 * 4 + dw;
    int k    = kk * 32 + ((lane >> 4) << 4) + 2 * v;
    int n    = nt * 16 + (lane & 15);

    float w0 = Wg[(size_t)k * NF + n];
    float w1 = Wg[(size_t)(k + 1) * NF + n];
    _Float16 h0 = (_Float16)w0, h1 = (_Float16)w1;
    _Float16 l0 = (_Float16)(w0 - (float)h0);
    _Float16 l1 = (_Float16)(w1 - (float)h1);
    union { _Float16 h[2]; unsigned int u; } ph, pl;
    ph.h[0] = h0; ph.h[1] = h1;
    pl.h[0] = l0; pl.h[1] = l1;
    whi[d] = ph.u;
    wlo[d] = pl.u;
}

// ---------------------------------------------------------------------------
// Fused scan: one block per batch element, 32 waves (wave32) = 1024 threads.
// ---------------------------------------------------------------------------
__launch_bounds__(1024, 1)
__global__ void cas_fused_scan(const float* __restrict__ xg,
                               const unsigned int* __restrict__ whig,
                               const unsigned int* __restrict__ wlog,
                               const float* __restrict__ Wcg,
                               const float* __restrict__ Wlg,
                               const float* __restrict__ Wlcg,
                               const float* __restrict__ blg,
                               const float* __restrict__ bg,
                               const float* __restrict__ ug,
                               float* __restrict__ out_att,
                               float* __restrict__ out_ctx,
                               float* __restrict__ out_wgt) {
    __shared__ __align__(16) unsigned int whi_lds[WFRAG_DW];   // 128 KB, loaded once
    __shared__ __align__(16) _Float16 xhi[NS * XPAD];          // 33 KB
    __shared__ __align__(16) _Float16 xlo[NS * XPAD];          // 33 KB
    __shared__ float ctx[NF];
    __shared__ float cvec[NF];
    __shared__ float part[4 * NF];
    __shared__ float lg[NS];
    __shared__ float aits[NS];
    __shared__ float aw[NS];
    __shared__ float wlsh[NF], wlcsh[NF];

    const int tid  = threadIdx.x;
    const int lane = tid & 31;
    const int wid  = tid >> 5;
    const int b    = blockIdx.x;
    const int jcol = tid & 255;
    const int iblk = tid >> 8;

    // Register-cache W_context (64 VGPRs): column jcol, rows iblk*64..+63.
    float wcreg[64];
#pragma unroll
    for (int ii = 0; ii < 64; ++ii)
        wcreg[ii] = Wcg[(size_t)(iblk * 64 + ii) * NF + jcol];

    // W_hi fragments -> LDS once (b128 copies, reused for all 128 steps).
    {
        const uint4* src = (const uint4*)whig;
        uint4* dst = (uint4*)whi_lds;
#pragma unroll
        for (int q = 0; q < 8; ++q) {
            int idx = tid + 1024 * q;       // 8192 uint4
            dst[idx] = src[idx];
        }
    }
    if (tid < NF) {
        ctx[tid]   = 0.0f;
        wlsh[tid]  = Wlg[tid];
        wlcsh[tid] = Wlcg[tid];
    }
    const float bl0 = blg[0];

    // WMMA tile mapping: 64 tiles (4 M x 16 N) over 32 waves -> 2 M-tiles/wave.
    const int nt   = wid & 15;
    const int m0   = (wid >> 4) * 32;      // rows m0..m0+15 and m0+16..m0+31
    const int m1   = m0 + 16;
    const int l15  = lane & 15;
    const int hi   = lane >> 4;
    const int fcol = nt * 16 + l15;        // this thread's output feature column
    const float bf_r = bg[fcol];           // bias/u for the fused epilogue
    const float uf_r = ug[fcol];
    __syncthreads();

    for (int t = 0; t < NT; ++t) {
        const float* xtile = xg + ((size_t)(b * NT + t)) * NS * NF;

        // Prefetch next step's x tile (1024 x 64B lines = 64 KB).
        if (t + 1 < NT) {
            const char* nx = (const char*)(xtile + NS * NF);
            __builtin_prefetch(nx + (size_t)tid * 64, 0, 0);
        }

        // Stage x tile as f16 hi/lo pairs (coalesced float2 loads, b32 LDS stores).
        {
            const float2* xt2 = (const float2*)xtile;
#pragma unroll
            for (int j = 0; j < 8; ++j) {
                int p = tid + 1024 * j;            // pair index, 8192 pairs
                float2 v = xt2[p];
                int row = p >> 7, cp = p & 127;
                _Float16 h0 = (_Float16)v.x, h1 = (_Float16)v.y;
                _Float16 l0 = (_Float16)(v.x - (float)h0);
                _Float16 l1 = (_Float16)(v.y - (float)h1);
                union { _Float16 h[2]; unsigned int u; } ph, pl;
                ph.h[0] = h0; ph.h[1] = h1;
                pl.h[0] = l0; pl.h[1] = l1;
                *(unsigned int*)&xhi[row * XPAD + 2 * cp] = ph.u;
                *(unsigned int*)&xlo[row * XPAD + 2 * cp] = pl.u;
            }
        }
        __syncthreads();

        // --- c = ctx @ W_context partials (register-cached) ---
        {
            float acc = 0.0f;
#pragma unroll
            for (int ii = 0; ii < 64; ++ii)
                acc += ctx[iblk * 64 + ii] * wcreg[ii];
            part[iblk * NF + jcol] = acc;
        }
        // --- gate logits: wave w owns sentences 2w, 2w+1 ---
        {
            const int s0 = 2 * wid, s1 = 2 * wid + 1;
            float gl = 0.0f, g0 = 0.0f, g1 = 0.0f;
#pragma unroll
            for (int j = 0; j < 4; ++j) {
                int f = 2 * lane + 64 * j;
                float wl0 = wlsh[f], wl1 = wlsh[f + 1];
                gl += ctx[f] * wlcsh[f] + ctx[f + 1] * wlcsh[f + 1];
                float xa0 = (float)xhi[s0 * XPAD + f]     + (float)xlo[s0 * XPAD + f];
                float xa1 = (float)xhi[s0 * XPAD + f + 1] + (float)xlo[s0 * XPAD + f + 1];
                float xb0 = (float)xhi[s1 * XPAD + f]     + (float)xlo[s1 * XPAD + f];
                float xb1 = (float)xhi[s1 * XPAD + f + 1] + (float)xlo[s1 * XPAD + f + 1];
                g0 += xa0 * wl0 + xa1 * wl1;
                g1 += xb0 * wl0 + xb1 * wl1;
            }
            gl = wred32(gl); g0 = wred32(g0); g1 = wred32(g1);
            if (lane == 0) {
                lg[s0] = 1.0f / (1.0f + expf(-(gl + bl0 + g0)));
                lg[s1] = 1.0f / (1.0f + expf(-(gl + bl0 + g1)));
            }
        }
        if (tid < NS) aits[tid] = 0.0f;
        __syncthreads();
        if (tid < NF)
            cvec[tid] = part[tid] + part[NF + tid] + part[2 * NF + tid] + part[3 * NF + tid];
        __syncthreads();

        // --- uit = x @ W via f16 3-term split WMMA, fused tanh/gate/dot(u) ---
        {
            v8f acc0 = {};
            v8f acc1 = {};
#pragma unroll 2
            for (int kk = 0; kk < 8; ++kk) {
                // B fragments: W_hi from LDS, W_lo streamed (coalesced b128).
                v16h bhi, blo;
                {
                    int base = ((kk * 16 + nt) * 2) * 32 + lane;   // *4 dwords
                    ((uint4*)&bhi)[0] = *(const uint4*)&whi_lds[(base) * 4];
                    ((uint4*)&bhi)[1] = *(const uint4*)&whi_lds[(base + 32) * 4];
                    ((uint4*)&blo)[0] = ((const uint4*)wlog)[base];
                    ((uint4*)&blo)[1] = ((const uint4*)wlog)[base + 32];
                }
                // A fragments (contiguous k order, same as B): v0-7 = K0+16h..K0+16h+15
                const int ko = 32 * kk + 16 * hi;
                {
                    v16h ah, al;
                    int r = (m0 + l15) * XPAD + ko;
                    ((uint4*)&ah)[0] = *(const uint4*)&xhi[r];
                    ((uint4*)&ah)[1] = *(const uint4*)&xhi[r + 8];
                    ((uint4*)&al)[0] = *(const uint4*)&xlo[r];
                    ((uint4*)&al)[1] = *(const uint4*)&xlo[r + 8];
                    acc0 = __builtin_amdgcn_wmma_f32_16x16x32_f16(false, ah, false, bhi,
                                                                  (short)0, acc0, false, false);
                    acc0 = __builtin_amdgcn_wmma_f32_16x16x32_f16(false, ah, false, blo,
                                                                  (short)0, acc0, false, false);
                    acc0 = __builtin_amdgcn_wmma_f32_16x16x32_f16(false, al, false, bhi,
                                                                  (short)0, acc0, false, false);
                }
                {
                    v16h ah, al;
                    int r = (m1 + l15) * XPAD + ko;
                    ((uint4*)&ah)[0] = *(const uint4*)&xhi[r];
                    ((uint4*)&ah)[1] = *(const uint4*)&xhi[r + 8];
                    ((uint4*)&al)[0] = *(const uint4*)&xlo[r];
                    ((uint4*)&al)[1] = *(const uint4*)&xlo[r + 8];
                    acc1 = __builtin_amdgcn_wmma_f32_16x16x32_f16(false, ah, false, bhi,
                                                                  (short)0, acc1, false, false);
                    acc1 = __builtin_amdgcn_wmma_f32_16x16x32_f16(false, ah, false, blo,
                                                                  (short)0, acc1, false, false);
                    acc1 = __builtin_amdgcn_wmma_f32_16x16x32_f16(false, al, false, bhi,
                                                                  (short)0, acc1, false, false);
                }
            }
            // Fused epilogue on C-fragments: VGPR r -> row r / r+8, lane -> column.
            const float cf = cvec[fcol];
#pragma unroll
            for (int r = 0; r < 8; ++r) {
                int sa = m0 + r + 8 * hi;
                float lga = lg[sa];
                float ha = tanhf(lga * acc0[r] + (1.0f - lga) * cf + bf_r);
                float pa = wred16(ha * uf_r);
                if (l15 == 0) atomicAdd(&aits[sa], pa);

                int sb = m1 + r + 8 * hi;
                float lgb = lg[sb];
                float hb = tanhf(lgb * acc1[r] + (1.0f - lgb) * cf + bf_r);
                float pb = wred16(hb * uf_r);
                if (l15 == 0) atomicAdd(&aits[sb], pb);
            }
        }
        __syncthreads();

        // --- softmax over sentences (reference form: exp / (sum + eps)) ---
        if (wid == 0) {
            float e0 = expf(aits[lane]);
            float e1 = expf(aits[lane + 32]);
            float s = wred32(e0 + e1);
            float inv = 1.0f / (s + EPSF);
            aw[lane]      = e0 * inv;
            aw[lane + 32] = e1 * inv;
            size_t wbase = ((size_t)(b * NT + t)) * NS;
            out_wgt[wbase + lane]      = e0 * inv;
            out_wgt[wbase + lane + 32] = e1 * inv;
        }
        __syncthreads();

        // --- attended = sum_s a[s] * x[s,:] ; context update ---
        {
            float acc = 0.0f;
#pragma unroll
            for (int ss = 0; ss < 16; ++ss) {
                int s = iblk * 16 + ss;
                float xv = (float)xhi[s * XPAD + jcol] + (float)xlo[s * XPAD + jcol];
                acc += aw[s] * xv;
            }
            part[iblk * NF + jcol] = acc;
        }
        __syncthreads();
        if (tid < NF) {
            float att = part[tid] + part[NF + tid] + part[2 * NF + tid] + part[3 * NF + tid];
            float nc = ctx[tid] + att;
            ctx[tid] = nc;
            size_t obase = ((size_t)(b * NT + t)) * NF;
            out_att[obase + tid] = att;
            out_ctx[obase + tid] = nc;
        }
        __syncthreads();
    }
}

extern "C" void kernel_launch(void* const* d_in, const int* in_sizes, int n_in,
                              void* d_out, int out_size, void* d_ws, size_t ws_size,
                              hipStream_t stream) {
    const float* x   = (const float*)d_in[0];
    const float* W   = (const float*)d_in[1];
    const float* Wc  = (const float*)d_in[2];
    const float* Wl  = (const float*)d_in[3];
    const float* Wlc = (const float*)d_in[4];
    const float* bl  = (const float*)d_in[5];
    const float* bb  = (const float*)d_in[6];
    const float* u   = (const float*)d_in[7];

    unsigned int* whi = (unsigned int*)d_ws;            // 128 KB
    unsigned int* wlo = whi + WFRAG_DW;                 // 128 KB

    float* out     = (float*)d_out;
    float* out_att = out;                               // [B,T,F]
    float* out_ctx = out + (size_t)NB * NT * NF;        // [B,T,F]
    float* out_wgt = out + (size_t)2 * NB * NT * NF;    // [B,T,S]

    prep_w_frags<<<dim3(WFRAG_DW / 256), dim3(256), 0, stream>>>(W, whi, wlo);
    cas_fused_scan<<<dim3(NB), dim3(1024), 0, stream>>>(
        x, whi, wlo, Wc, Wl, Wlc, bl, bb, u, out_att, out_ctx, out_wgt);
}
